// MultiQueryAttention_45543833206995
// MI455X (gfx1250) — matrix operand order repaired and checked by reading, not compiled, not run on hardware
//
#include <hip/hip_runtime.h>
#include <hip/hip_bf16.h>

// ---------------------------------------------------------------------------
// MQA attention block for MI455X (gfx1250).
// All matmuls via v_wmma_f32_16x16x32_bf16; global->LDS staging via
// GLOBAL_LOAD_ASYNC_TO_LDS_B128 (ASYNCcnt) with double buffering; softmax
// row reductions via DPP16 (no LDS bpermute round trips).
// ---------------------------------------------------------------------------

typedef __attribute__((ext_vector_type(16))) __bf16 v16bf;
typedef __attribute__((ext_vector_type(4)))  __bf16 v4bf;
typedef __attribute__((ext_vector_type(8)))  float  v8f;

#define D_MODEL  2048
#define N_HEADS  16
#define HEAD_DIM 128
#define BATCH    2
#define SEQ      2048
#define MTOT     (BATCH * SEQ)   // 4096 rows total

static __device__ inline v8f wmma_bf16(v16bf a, v16bf b, v8f c) {
  // D = A(16x32 bf16) x B(32x16 bf16) + C(16x16 f32)
  return __builtin_amdgcn_wmma_f32_16x16x32_bf16(
      /*neg_a=*/false, a, /*neg_b=*/false, b,
      /*c_mod=*/(short)0, c, /*reuse_a=*/false, /*reuse_b=*/false);
}

static __device__ inline v8f zero8() {
  v8f z;
#pragma unroll
  for (int i = 0; i < 8; ++i) z[i] = 0.0f;
  return z;
}

// Low 32 bits of a generic pointer into __shared__ = wave-relative LDS offset,
// which is what the async-to-LDS instruction's VDST operand expects.
static __device__ inline unsigned lds_off(const void* p) {
  return (unsigned)(unsigned long long)p;
}

// CDNA5 async copy: 16B per lane, global -> LDS, tracked by ASYNCcnt.
static __device__ inline void async_load_b128(unsigned lds_addr,
                                              const void* gaddr) {
  asm volatile("global_load_async_to_lds_b128 %0, %1, off"
               :: "v"(lds_addr), "v"(gaddr)
               : "memory");
}

static __device__ inline void wait_async0() {
  asm volatile("s_wait_asynccnt 0x0" ::: "memory");
}

// ---- DPP16 16-lane (half-wave) reductions: pure VALU, co-issues with WMMA ---
template <int CTRL>
static __device__ inline float dpp16_f(float x) {
  return __int_as_float(__builtin_amdgcn_update_dpp(
      0, __float_as_int(x), CTRL, /*row_mask=*/0xF, /*bank_mask=*/0xF,
      /*bound_ctrl=*/true));
}
static __device__ inline float row16_max(float v) {
  v = fmaxf(v, dpp16_f<0xB1>(v));   // quad_perm(1,0,3,2): xor 1
  v = fmaxf(v, dpp16_f<0x4E>(v));   // quad_perm(2,3,0,1): xor 2
  v = fmaxf(v, dpp16_f<0x141>(v));  // row_half_mirror:    xor 7 -> quads
  v = fmaxf(v, dpp16_f<0x140>(v));  // row_mirror:         xor 15 -> row
  return v;
}
static __device__ inline float row16_sum(float v) {
  v += dpp16_f<0xB1>(v);
  v += dpp16_f<0x4E>(v);
  v += dpp16_f<0x141>(v);
  v += dpp16_f<0x140>(v);
  return v;
}

// ---------------------------------------------------------------------------
// fp32 -> bf16 conversion (vectorized 4-wide)
// ---------------------------------------------------------------------------
__global__ void cvt_f32_bf16(const float4* __restrict__ src,
                             __bf16* __restrict__ dst, int n4) {
  int i = blockIdx.x * blockDim.x + threadIdx.x;
  const int stride = gridDim.x * blockDim.x;
  for (; i < n4; i += stride) {
    const float4 v = src[i];
    v4bf o;
    o[0] = (__bf16)v.x; o[1] = (__bf16)v.y;
    o[2] = (__bf16)v.z; o[3] = (__bf16)v.w;
    *reinterpret_cast<v4bf*>(&dst[4 * i]) = o;
  }
}

// ---------------------------------------------------------------------------
// GEMM: Y[M,N] = Xbf[M,K] @ Wbf[N,K]^T + bias[N]
// 256 threads (8 waves). Block tile 128(M) x 128(N); wave tile 32x64
// (2x4 WMMA tiles -> 8 WMMAs per 32-K panel). Double-buffered async staging.
// ---------------------------------------------------------------------------
template <typename OutT>
__global__ __launch_bounds__(256) void gemm_bias_bf16(
    const __bf16* __restrict__ X, const __bf16* __restrict__ W,
    const float* __restrict__ bias, OutT* __restrict__ Y,
    int M, int N, int K) {
  __shared__ __bf16 Xs[2][128 * 32];
  __shared__ __bf16 Ws[2][128 * 32];

  const int tid  = threadIdx.x;
  const int lane = tid & 31;
  const int wave = tid >> 5;
  const int m0 = blockIdx.y * 128;
  const int n0 = blockIdx.x * 128;
  const int mBase = (wave & 3) * 32;    // 4 waves along M
  const int nBase = (wave >> 2) * 64;   // 2 waves along N

  const int lm  = lane & 15;   // A: row m / B,C,D: col n
  const int hi  = lane >> 4;   // half-wave select
  const int kbA = hi * 8;      // unused (A frags use kbB)
  const int kbB = hi * 16;     // B frag K base

  v8f acc[2][4];
#pragma unroll
  for (int i = 0; i < 2; ++i)
#pragma unroll
    for (int j = 0; j < 4; ++j) acc[i][j] = zero8();

  const int xr = tid >> 2;        // 0..63: staging row within a 64-row pass
  const int xc = (tid & 3) * 8;   // 8 bf16 (16B) per thread per row

  const int nk = K >> 5;

  // async-stage one 32-wide K panel of X (128x32) and W (128x32)
  auto issue_panel = [&](int kb, int buf) {
    const int k0 = kb << 5;
#pragma unroll
    for (int p = 0; p < 2; ++p) {
      const int row = p * 64 + xr;
      async_load_b128(lds_off(&Xs[buf][row * 32 + xc]),
                      &X[(size_t)(m0 + row) * K + k0 + xc]);
      async_load_b128(lds_off(&Ws[buf][row * 32 + xc]),
                      &W[(size_t)(n0 + row) * K + k0 + xc]);
    }
  };

  issue_panel(0, 0);
  for (int kb = 0; kb < nk; ++kb) {
    wait_async0();        // this wave's panel kb has landed in LDS
    __syncthreads();      // every wave's panel kb landed; prior reads done
    if (kb + 1 < nk) issue_panel(kb + 1, (kb + 1) & 1);  // overlap with WMMAs

    const __bf16* Xb = Xs[kb & 1];
    const __bf16* Wb = Ws[kb & 1];

    // A fragments (16x32): lane lm holds row m=lm, K={kbB..+15}
    v16bf a[2];
#pragma unroll
    for (int mi = 0; mi < 2; ++mi) {
      const __bf16* p = &Xb[(mBase + mi * 16 + lm) * 32];
#pragma unroll
      for (int i = 0; i < 8; ++i) { a[mi][i] = p[kbB + i]; a[mi][8 + i] = p[kbB + 8 + i]; }
    }
    // B fragments (32x16): element i -> (K=kbB+i, N=lm); B(k,n) = W[n][k]
    v16bf b[4];
#pragma unroll
    for (int ni = 0; ni < 4; ++ni) {
      const __bf16* p = &Wb[(nBase + ni * 16 + lm) * 32];
#pragma unroll
      for (int i = 0; i < 16; ++i) b[ni][i] = p[kbB + i];
    }
#pragma unroll
    for (int mi = 0; mi < 2; ++mi)
#pragma unroll
      for (int ni = 0; ni < 4; ++ni)
        acc[mi][ni] = wmma_bf16(a[mi], b[ni], acc[mi][ni]);
  }

  // C/D layout: element r of lane -> row = r + 8*hi, col = lm
#pragma unroll
  for (int mi = 0; mi < 2; ++mi) {
#pragma unroll
    for (int ni = 0; ni < 4; ++ni) {
      const int col = n0 + nBase + ni * 16 + lm;
      const float bv = bias[col];
#pragma unroll
      for (int r = 0; r < 8; ++r) {
        const int row = m0 + mBase + mi * 16 + r + hi * 8;
        const float v = acc[mi][ni][r] + bv;
        Y[(size_t)row * N + col] = (OutT)v;
      }
    }
  }
}

// ---------------------------------------------------------------------------
// MQA flash attention: grid (B*H, SEQ/128), 256 threads (8 waves).
// Wave w owns 16 query rows q0blk + 16*w. K/V tiles (32x128 bf16) shared in
// LDS across all waves (single KV head), double-buffered async staging.
// Online softmax in registers with DPP16 row reductions; P re-laid out
// C->A via per-wave LDS scratch.
// ---------------------------------------------------------------------------
__global__ __launch_bounds__(256) void mqa_attn(
    const __bf16* __restrict__ Q, const __bf16* __restrict__ Kc,
    const __bf16* __restrict__ Vc, __bf16* __restrict__ Aout) {
  __shared__ __bf16 Ks[2][32 * 128];
  __shared__ __bf16 Vs[2][32 * 128];
  __shared__ __bf16 Ps[8 * 16 * 32];  // per-wave P scratch

  const int tid  = threadIdx.x;
  const int lane = tid & 31;
  const int wave = tid >> 5;
  const int b = blockIdx.x / N_HEADS;
  const int h = blockIdx.x % N_HEADS;
  const int q0blk = blockIdx.y * 128;
  const int q0w   = q0blk + wave * 16;

  const int lm  = lane & 15;
  const int hi  = lane >> 4;
  const int kbA = hi * 8;
  const int kbB = hi * 16;
  const float scale = 0.08838834764831845f;  // 1/sqrt(128)

  // Q fragments for all 128 head dims (4 K-steps of 32)
  v16bf qf[4];
#pragma unroll
  for (int kk = 0; kk < 4; ++kk) {
    const __bf16* qp =
        Q + (size_t)(b * SEQ + q0w + lm) * D_MODEL + h * HEAD_DIM + kk * 32;
#pragma unroll
    for (int i = 0; i < 8; ++i) { qf[kk][i] = qp[kbB + i]; qf[kk][8 + i] = qp[kbB + 8 + i]; }
  }

  v8f oacc[8];
#pragma unroll
  for (int i = 0; i < 8; ++i) oacc[i] = zero8();
  float m_run[8], l_run[8];
#pragma unroll
  for (int r = 0; r < 8; ++r) { m_run[r] = -1e30f; l_run[r] = 0.0f; }

  const int sr  = tid >> 3;        // 0..31: K/V tile row
  const int scg = (tid & 7) * 16;  // 16 bf16 (2x16B) per thread per row

  const int nIter = (q0blk + 128) >> 5;  // key blocks of 32

  auto issue_kv = [&](int it, int buf) {
    const size_t grow = (size_t)(b * SEQ + (it << 5) + sr) * HEAD_DIM + scg;
    async_load_b128(lds_off(&Ks[buf][sr * 128 + scg]),     &Kc[grow]);
    async_load_b128(lds_off(&Ks[buf][sr * 128 + scg + 8]), &Kc[grow + 8]);
    async_load_b128(lds_off(&Vs[buf][sr * 128 + scg]),     &Vc[grow]);
    async_load_b128(lds_off(&Vs[buf][sr * 128 + scg + 8]), &Vc[grow + 8]);
  };

  issue_kv(0, 0);
  for (int it = 0; it < nIter; ++it) {
    const int s0 = it << 5;
    wait_async0();
    __syncthreads();
    if (it + 1 < nIter) issue_kv(it + 1, (it + 1) & 1);  // overlap with compute

    const __bf16* Kb = Ks[it & 1];
    const __bf16* Vb = Vs[it & 1];

    // scores S = Q K^T : 16x32 tile as two 16x16 accumulators
    v8f sc[2];
#pragma unroll
    for (int j = 0; j < 2; ++j) {
      v8f s = zero8();
#pragma unroll
      for (int kk = 0; kk < 4; ++kk) {
        v16bf bfr;  // B(k,n) = K[key j*16+n][dim kk*32+k]
        const __bf16* kp = &Kb[(j * 16 + lm) * 128 + kk * 32 + kbB];
#pragma unroll
        for (int i = 0; i < 16; ++i) bfr[i] = kp[i];
        s = wmma_bf16(qf[kk], bfr, s);
      }
      sc[j] = s;
    }

    // online softmax (strict causal: key < query) + P -> LDS (bf16)
    __bf16* pp = &Ps[wave * 512];
#pragma unroll
    for (int r = 0; r < 8; ++r) {
      const int rowm = r + hi * 8;
      const int qrow = q0w + rowm;
      const float s0v = ((s0 + lm) < qrow)      ? sc[0][r] * scale : -1e30f;
      const float s1v = ((s0 + 16 + lm) < qrow) ? sc[1][r] * scale : -1e30f;
      const float tmax = row16_max(fmaxf(s0v, s1v));
      const float mprev = m_run[r];
      const float mnew  = fmaxf(mprev, tmax);
      const float p0 = __expf(s0v - mnew);
      const float p1 = __expf(s1v - mnew);
      const float rs = row16_sum(p0 + p1);
      const float alpha = __expf(mprev - mnew);
      m_run[r] = mnew;
      l_run[r] = l_run[r] * alpha + rs;
#pragma unroll
      for (int nd = 0; nd < 8; ++nd) oacc[nd][r] *= alpha;
      pp[rowm * 32 + lm]      = (__bf16)p0;
      pp[rowm * 32 + 16 + lm] = (__bf16)p1;
    }

    // P as A fragment (16x32); LDS ops from the same wave are in-order
    v16bf pf;
    {
      const __bf16* pr = &Ps[wave * 512 + lm * 32];
#pragma unroll
      for (int i = 0; i < 8; ++i) { pf[i] = pr[kbB + i]; pf[8 + i] = pr[kbB + 8 + i]; }
    }
    // O += P @ V : 8 head-dim tiles of 16
#pragma unroll
    for (int nd = 0; nd < 8; ++nd) {
      v16bf vfr;  // B(k,n) = V[key k][dim nd*16+n]
#pragma unroll
      for (int i = 0; i < 16; ++i) vfr[i] = Vb[(kbB + i) * 128 + nd * 16 + lm];
      oacc[nd] = wmma_bf16(pf, vfr, oacc[nd]);
    }
  }

  // normalize, zero row t==0 (no valid keys), store head-interleaved bf16
#pragma unroll
  for (int nd = 0; nd < 8; ++nd) {
#pragma unroll
    for (int r = 0; r < 8; ++r) {
      const int rowm = r + hi * 8;
      const int t = q0w + rowm;
      const float v = (t == 0) ? 0.0f : oacc[nd][r] / l_run[r];
      Aout[(size_t)(b * SEQ + t) * D_MODEL + h * HEAD_DIM + nd * 16 + lm] =
          (__bf16)v;
    }
  }
}

// ---------------------------------------------------------------------------
// Host launcher
// ---------------------------------------------------------------------------
extern "C" void kernel_launch(void* const* d_in, const int* in_sizes, int n_in,
                              void* d_out, int out_size, void* d_ws,
                              size_t ws_size, hipStream_t stream) {
  (void)in_sizes; (void)n_in; (void)out_size; (void)ws_size;
  const float* x  = (const float*)d_in[0];
  const float* Wq = (const float*)d_in[1];
  const float* bq = (const float*)d_in[2];
  const float* Wk = (const float*)d_in[3];
  const float* bk = (const float*)d_in[4];
  const float* Wv = (const float*)d_in[5];
  const float* bv = (const float*)d_in[6];
  const float* Wo = (const float*)d_in[7];
  const float* bo = (const float*)d_in[8];
  float* out = (float*)d_out;

  char* ws = (char*)d_ws;
  size_t off = 0;
  auto alloc = [&](size_t bytes) {
    void* p = ws + off;
    off += (bytes + 255) & ~(size_t)255;
    return p;
  };
  __bf16* Xbf = (__bf16*)alloc((size_t)MTOT * D_MODEL * 2);     // 16 MB
  __bf16* Wqb = (__bf16*)alloc((size_t)D_MODEL * D_MODEL * 2);  // 8 MB
  __bf16* Wkb = (__bf16*)alloc((size_t)HEAD_DIM * D_MODEL * 2);
  __bf16* Wvb = (__bf16*)alloc((size_t)HEAD_DIM * D_MODEL * 2);
  __bf16* Wob = (__bf16*)alloc((size_t)D_MODEL * D_MODEL * 2);  // 8 MB
  __bf16* Qb  = (__bf16*)alloc((size_t)MTOT * D_MODEL * 2);     // 16 MB
  __bf16* Kb  = (__bf16*)alloc((size_t)MTOT * HEAD_DIM * 2);
  __bf16* Vb  = (__bf16*)alloc((size_t)MTOT * HEAD_DIM * 2);
  __bf16* Ab  = (__bf16*)alloc((size_t)MTOT * D_MODEL * 2);     // 16 MB

  auto cvt = [&](const float* s, __bf16* d, size_t n) {
    cvt_f32_bf16<<<dim3(1024), dim3(256), 0, stream>>>(
        (const float4*)s, d, (int)(n / 4));
  };
  cvt(x,  Xbf, (size_t)MTOT * D_MODEL);
  cvt(Wq, Wqb, (size_t)D_MODEL * D_MODEL);
  cvt(Wk, Wkb, (size_t)HEAD_DIM * D_MODEL);
  cvt(Wv, Wvb, (size_t)HEAD_DIM * D_MODEL);
  cvt(Wo, Wob, (size_t)D_MODEL * D_MODEL);

  // projections (N=2048 for Q, N=128 for K/V -> grid.x 16 / 1)
  gemm_bias_bf16<__bf16><<<dim3(D_MODEL / 128, MTOT / 128), 256, 0, stream>>>(
      Xbf, Wqb, bq, Qb, MTOT, D_MODEL, D_MODEL);
  gemm_bias_bf16<__bf16><<<dim3(HEAD_DIM / 128, MTOT / 128), 256, 0, stream>>>(
      Xbf, Wkb, bk, Kb, MTOT, HEAD_DIM, D_MODEL);
  gemm_bias_bf16<__bf16><<<dim3(HEAD_DIM / 128, MTOT / 128), 256, 0, stream>>>(
      Xbf, Wvb, bv, Vb, MTOT, HEAD_DIM, D_MODEL);

  // attention
  mqa_attn<<<dim3(BATCH * N_HEADS, SEQ / 128), 256, 0, stream>>>(Qb, Kb, Vb, Ab);

  // output projection (fp32 out)
  gemm_bias_bf16<float><<<dim3(D_MODEL / 128, MTOT / 128), 256, 0, stream>>>(
      Ab, Wob, bo, out, MTOT, D_MODEL, D_MODEL);
}
